// InstanceGatherer_53635551592518
// MI455X (gfx1250) — matrix operand order repaired and checked by reading, not compiled, not run on hardware
//
#include <hip/hip_runtime.h>
#include <hip/hip_bf16.h>
#include <stdint.h>

typedef __attribute__((ext_vector_type(16))) __bf16 v16bf;
typedef __attribute__((ext_vector_type(8)))  float  v8f;

union FragBF { v16bf v; uint32_t u[8]; };

__device__ __forceinline__ unsigned short f2bf(float f) {
    union { float f; uint32_t u; } x; x.f = f;
    uint32_t r = x.u + 0x7FFFu + ((x.u >> 16) & 1u);   // RNE
    return (unsigned short)(r >> 16);
}
__device__ __forceinline__ float bf2f(unsigned short h) {
    return __uint_as_float(((uint32_t)h) << 16);
}

// A fragment: 16x32 bf16, LDS layout [row][k], stride 32 bf16.
// ISA table: lane<16: m=lane, K in {0..7, 16..23}; lane>=16: m=lane-16, K in {8..15, 24..31}
__device__ __forceinline__ v16bf frag_a(const unsigned short* lds, int row0, int lane) {
    int half = lane >> 4, m = lane & 15;
    const uint32_t* p = (const uint32_t*)(lds + (row0 + m) * 32);
    FragBF f;
#pragma unroll
    for (int v = 0; v < 8; ++v) {
        int kd = (v & 3) + ((v >> 2) << 3) + (half << 2); // dword index = k/2
        f.u[v] = p[kd];
    }
    return f.v;
}
// B fragment: 32x16 bf16, LDS layout transposed [col][k], stride 32 bf16.
// lane<16: n=lane, K in {0..7, 16..23}; lane>=16: n=lane-16, K in {8..15, 24..31} (2 bf16 per VGPR)
__device__ __forceinline__ v16bf frag_b(const unsigned short* lds, int col0, int lane) {
    int half = lane >> 4, n = lane & 15;
    const uint32_t* p = (const uint32_t*)(lds + (col0 + n) * 32);
    FragBF f;
#pragma unroll
    for (int v = 0; v < 8; ++v) f.u[v] = p[(v & 3) + ((v >> 2) << 3) + (half << 2)];
    return f.v;
}

// ---------------- K0: r[d][h] = sum_j Wk[d][h*32+j] * q[h*32+j],  qb[h] = q_h . bk_h
__global__ __launch_bounds__(256) void k0_prep(
    const float* __restrict__ pq, const float* __restrict__ Wq, const float* __restrict__ bq,
    const float* __restrict__ Wk, const float* __restrict__ bk,
    float* __restrict__ r_out, float* __restrict__ qb_out)
{
    __shared__ float ql[256];
    int t = threadIdx.x;
    float q = bq[t];
    for (int d = 0; d < 256; ++d) q += pq[d] * Wq[d * 256 + t];
    ql[t] = q;
    __syncthreads();
#pragma unroll
    for (int h = 0; h < 8; ++h) {
        float a = 0.f;
        for (int j = 0; j < 32; ++j) a += Wk[t * 256 + h * 32 + j] * ql[h * 32 + j];
        r_out[t * 8 + h] = a;
    }
    if (t < 8) {
        float a = 0.f;
        for (int j = 0; j < 32; ++j) a += bk[t * 32 + j] * ql[t * 32 + j];
        qb_out[t] = a;
    }
}

// ---------------- K1: tp = LN(tokens @ W_tok + b_tok) -> bf16 [32768 x 256]
__global__ __launch_bounds__(256) void k1_gemm_ln(
    const float* __restrict__ tokens, const float* __restrict__ Wt,
    const float* __restrict__ btok, const float* __restrict__ g,
    const float* __restrict__ be, unsigned short* __restrict__ tp16)
{
    __shared__ unsigned short As[32 * 32];    // 2KB  A tile [row][k]
    __shared__ unsigned short Bs[256 * 32];   // 16KB B tile transposed [col][k]
    __shared__ float Ep[32 * 256];            // 32KB epilogue
    __shared__ float mean_s[32], rstd_s[32];
    __shared__ float redA[256], redB[256];

    const int t = threadIdx.x, lane = t & 31, wave = t >> 5;
    const int wr = (wave & 1) * 16, wc = (wave >> 1) * 64;
    const int rowBase = blockIdx.x * 32;

    v8f acc[4] = { {}, {}, {}, {} };
    const int ar = t >> 3, ac = (t & 7) * 4;
    const float* arow = tokens + (size_t)(rowBase + ar) * 2048 + ac;

    for (int k0 = 0; k0 < 2048; k0 += 32) {
        __syncthreads();
        if (k0 + 32 < 2048) __builtin_prefetch(arow + k0 + 32, 0, 0); // next A tile
#pragma unroll
        for (int i = 0; i < 4; ++i) As[ar * 32 + ac + i] = f2bf(arow[k0 + i]);
#pragma unroll
        for (int kk = 0; kk < 32; ++kk) Bs[t * 32 + kk] = f2bf(Wt[(size_t)(k0 + kk) * 256 + t]);
        __syncthreads();
        v16bf af = frag_a(As, wr, lane);
#pragma unroll
        for (int cg = 0; cg < 4; ++cg) {
            v16bf bfv = frag_b(Bs, wc + cg * 16, lane);
            acc[cg] = __builtin_amdgcn_wmma_f32_16x16x32_bf16(
                false, af, false, bfv, (short)0, acc[cg], false, false);
        }
    }
    __syncthreads();
    { // dump accumulators + bias
        int mh = 8 * (lane >> 4), n0 = lane & 15;
#pragma unroll
        for (int cg = 0; cg < 4; ++cg) {
            int n = wc + cg * 16 + n0;
            float bias = btok[n];
#pragma unroll
            for (int v = 0; v < 8; ++v) Ep[(wr + mh + v) * 256 + n] = acc[cg][v] + bias;
        }
    }
    __syncthreads();
    { // LN stats: 8 threads per row
        int row = t >> 3, seg = t & 7;
        float s = 0.f, q = 0.f;
        const float* rp = Ep + row * 256 + seg * 32;
        for (int i = 0; i < 32; ++i) { float x = rp[i]; s += x; q += x * x; }
        redA[t] = s; redB[t] = q;
        __syncthreads();
        if (seg == 0) {
            float ss = 0.f, qq = 0.f;
            for (int j = 0; j < 8; ++j) { ss += redA[t + j]; qq += redB[t + j]; }
            float mean = ss * (1.f / 256.f);
            float var = qq * (1.f / 256.f) - mean * mean;
            mean_s[row] = mean;
            rstd_s[row] = rsqrtf(var + 1e-5f);
        }
    }
    __syncthreads();
    {
        float gc = g[t], bec = be[t];
        unsigned short* orow = tp16 + (size_t)rowBase * 256 + t;
        for (int j = 0; j < 32; ++j) {
            float v = (Ep[j * 256 + t] - mean_s[j]) * rstd_s[j] * gc + bec;
            orow[(size_t)j * 256] = f2bf(v);
        }
    }
}

// ---------------- K2: per (b,o) radix-select top-128 index set + 1/sum(weights)
__global__ __launch_bounds__(256) void k2_topk(
    const float* __restrict__ act, int* __restrict__ idxo, float* __restrict__ wrecip)
{
    __shared__ uint32_t keys[4096];
    __shared__ float redf[256];
    __shared__ int cnt, c1, c2;
    int bo = blockIdx.x, t = threadIdx.x;
    const float* row = act + (size_t)bo * 4096;
    float s = 0.f;
    for (int i = 0; i < 16; ++i) {
        int j = t + 256 * i;
        float f = row[j];
        s += f;
        uint32_t u = __float_as_uint(f);
        keys[j] = u ^ ((u >> 31) ? 0xFFFFFFFFu : 0x80000000u); // order-preserving
    }
    redf[t] = s; __syncthreads();
    for (int st = 128; st; st >>= 1) { if (t < st) redf[t] += redf[t + st]; __syncthreads(); }
    if (t == 0) wrecip[bo] = 1.0f / fmaxf(redf[0], 1e-8f);

    uint32_t prefix = 0;
    for (int bit = 31; bit >= 0; --bit) {
        uint32_t cand = prefix | (1u << bit);
        if (t == 0) cnt = 0;
        __syncthreads();
        int lc = 0;
        for (int i = 0; i < 16; ++i) lc += (keys[t + 256 * i] >= cand);
        if (lc) atomicAdd(&cnt, lc);
        __syncthreads();
        if (cnt >= 128) prefix = cand;
        __syncthreads();
    }
    if (t == 0) cnt = 0;
    __syncthreads();
    { int lc = 0; for (int i = 0; i < 16; ++i) lc += (keys[t + 256 * i] > prefix);
      if (lc) atomicAdd(&cnt, lc); }
    __syncthreads();
    int nG = cnt;
    if (t == 0) { c1 = 0; c2 = nG; }
    __syncthreads();
    int* out = idxo + bo * 128;
    for (int i = 0; i < 16; ++i) {
        int j = t + 256 * i;
        uint32_t k = keys[j];
        if (k > prefix) { int p = atomicAdd(&c1, 1); out[p] = j; }
        else if (k == prefix) { int p = atomicAdd(&c2, 1); if (p < 128) out[p] = j; }
    }
}

// ---------------- K3: weighted_avg = (act/sum) @ tp  per batch, WMMA bf16
__global__ __launch_bounds__(128) void k3_wavg(
    const float* __restrict__ act, const float* __restrict__ wrecip,
    const unsigned short* __restrict__ tp16, float* __restrict__ wavg)
{
    __shared__ unsigned short As[16 * 32];  // [row][k]
    __shared__ unsigned short Bs[64 * 32];  // [col][k]
    int t = threadIdx.x, lane = t & 31, wave = t >> 5;
    int o0 = blockIdx.x * 16, c0 = blockIdx.y * 64, b = blockIdx.z;

    v8f acc = {};
    int ar = t >> 3, ac = (t & 7) * 4;
    int o = o0 + ar;
    int oc = (o < 100) ? o : 99;
    float wscale = (o < 100) ? wrecip[b * 100 + oc] : 0.0f;
    const float* arow = act + (size_t)(b * 100 + oc) * 4096 + ac;
    int bcol = t & 63, kh = (t >> 6) * 16;

    for (int k0 = 0; k0 < 4096; k0 += 32) {
        __syncthreads();
        if (k0 + 32 < 4096) __builtin_prefetch(arow + k0 + 32, 0, 0);
#pragma unroll
        for (int i = 0; i < 4; ++i) As[ar * 32 + ac + i] = f2bf(arow[k0 + i] * wscale);
#pragma unroll
        for (int j = 0; j < 16; ++j)
            Bs[bcol * 32 + kh + j] = tp16[(size_t)(b * 4096 + k0 + kh + j) * 256 + c0 + bcol];
        __syncthreads();
        v16bf af = frag_a(As, 0, lane);
        v16bf bfv = frag_b(Bs, wave * 16, lane);
        acc = __builtin_amdgcn_wmma_f32_16x16x32_bf16(
            false, af, false, bfv, (short)0, acc, false, false);
    }
    int mh = 8 * (lane >> 4), n0 = lane & 15;
#pragma unroll
    for (int v = 0; v < 8; ++v) {
        int oo = o0 + mh + v;
        if (oo < 100)
            wavg[(size_t)(b * 100 + oo) * 256 + c0 + wave * 16 + n0] = acc[v];
    }
}

// Async gather of one 128-row x 64-bf16 chunk of selected tp rows into LDS.
// Uses GLOBAL_LOAD_ASYNC_TO_LDS_B128 (memory -> LDS, no VGPR staging, ASYNCcnt).
// tpw_b points at batch base (row stride 128 dwords). Caller must __syncthreads()
// after this returns (s_wait_asynccnt only covers this wave's copies).
__device__ __forceinline__ void gather_chunk_async(
    const uint32_t* tpw_b, const int* ids, uint32_t* Xc, int c, int t)
{
    uint32_t lds_base = (uint32_t)(uintptr_t)Xc;   // low 32 bits of flat ptr = LDS addr
#pragma unroll
    for (int i = 0; i < 4; ++i) {
        int lin = t + 256 * i;                     // 1024 x 16-byte units
        int row = lin >> 3, unit = lin & 7;        // 8 units per 128B row-chunk
        uint64_t ga = (uint64_t)(uintptr_t)(tpw_b + (size_t)ids[row] * 128)
                    + (uint64_t)(c * 128 + unit * 16);
        uint32_t la = lds_base + (uint32_t)(lin * 16);
        asm volatile("global_load_async_to_lds_b128 %0, %1, off"
                     :: "v"(la), "v"(ga) : "memory");
    }
    asm volatile("s_wait_asynccnt 0" ::: "memory");
}

// ---------------- K4: gather + collapsed attention + Wo + 2x LN -> out
__global__ __launch_bounds__(256) void k4_attn(
    const unsigned short* __restrict__ tp16, const int* __restrict__ idx,
    const float* __restrict__ r, const float* __restrict__ qb,
    const float* __restrict__ Wv, const float* __restrict__ bv,
    const float* __restrict__ Wo, const float* __restrict__ b_o,
    const float* __restrict__ g_pn, const float* __restrict__ bt_pn,
    const float* __restrict__ g_on, const float* __restrict__ bt_on,
    const float* __restrict__ wavg, float* __restrict__ out)
{
    __shared__ int ids[128];
    __shared__ uint32_t Xc[128 * 32];   // 16KB: 128 rows x 64 bf16 chunk
    __shared__ float rl[2048];          // 8KB  staged r[256][8]
    __shared__ float qbl[8];
    __shared__ float sc[8 * 128];
    __shared__ float attn[8 * 128];
    __shared__ float Yl[8 * 256];
    __shared__ float pl[256];
    __shared__ float redA[256], redB[256];

    int boi = blockIdx.x;
    int b = boi / 100;
    int t = threadIdx.x, lane = t & 31, wave = t >> 5;
    if (t < 128) ids[t] = idx[boi * 128 + t];
    for (int i = t; i < 2048; i += 256) rl[i] = r[i];
    if (t < 8) qbl[t] = qb[t];
    __syncthreads();

    const uint32_t* tpw_b = (const uint32_t*)tp16 + (size_t)b * 4096 * 128;
    const int h_ = t & 7;
    float accs[4] = { 0.f, 0.f, 0.f, 0.f };

    // scores = Xg @ r, chunked over D
    for (int c = 0; c < 4; ++c) {
        __syncthreads();
        gather_chunk_async(tpw_b, ids, Xc, c, t);
        __syncthreads();
        const unsigned short* X = (const unsigned short*)Xc;
#pragma unroll
        for (int i = 0; i < 4; ++i) {
            int kk = (t >> 3) + 32 * i;
            const unsigned short* xr = X + kk * 64;
            float s = 0.f;
            for (int d = 0; d < 64; ++d) s += bf2f(xr[d]) * rl[(c * 64 + d) * 8 + h_];
            accs[i] += s;
        }
    }
#pragma unroll
    for (int i = 0; i < 4; ++i) {
        int kk = (t >> 3) + 32 * i;
        sc[h_ * 128 + kk] = (accs[i] + qbl[h_]) * 0.17677669529663687f; // 1/sqrt(32)
    }
    __syncthreads();

    // softmax: one wave per head, shuffle reductions
    {
        float m = -3.4e38f;
#pragma unroll
        for (int j = 0; j < 4; ++j) m = fmaxf(m, sc[wave * 128 + lane + 32 * j]);
        for (int d = 16; d; d >>= 1) m = fmaxf(m, __shfl_xor(m, d, 32));
        float s = 0.f, e[4];
#pragma unroll
        for (int j = 0; j < 4; ++j) { e[j] = expf(sc[wave * 128 + lane + 32 * j] - m); s += e[j]; }
        for (int d = 16; d; d >>= 1) s += __shfl_xor(s, d, 32);
        float inv = 1.0f / s;
#pragma unroll
        for (int j = 0; j < 4; ++j) attn[wave * 128 + lane + 32 * j] = e[j] * inv;
    }
    __syncthreads();

    // Y[h][d] = attn_h^T Xg
    for (int c = 0; c < 4; ++c) {
        __syncthreads();
        gather_chunk_async(tpw_b, ids, Xc, c, t);
        __syncthreads();
        const unsigned short* X = (const unsigned short*)Xc;
#pragma unroll
        for (int i = 0; i < 2; ++i) {
            int oi = t + 256 * i;
            int h = oi >> 6, d = oi & 63;
            float y = 0.f;
            for (int k = 0; k < 128; ++k) y += attn[h * 128 + k] * bf2f(X[k * 64 + d]);
            Yl[h * 256 + c * 64 + d] = y;
        }
    }
    __syncthreads();

    // pooled[j] = Y[h] @ Wv[:, j] + bv[j]   (h = j/32)
    {
        int h = t >> 5;
        float p = bv[t];
        for (int d = 0; d < 256; ++d) p += Yl[h * 256 + d] * Wv[d * 256 + t];
        pl[t] = p;
    }
    __syncthreads();
    // u = pooled @ Wo + bo
    float u = b_o[t];
    for (int d = 0; d < 256; ++d) u += pl[d] * Wo[d * 256 + t];

    // LN1
    redA[t] = u; redB[t] = u * u; __syncthreads();
    for (int st = 128; st; st >>= 1) { if (t < st) { redA[t] += redA[t + st]; redB[t] += redB[t + st]; } __syncthreads(); }
    float m1 = redA[0] * (1.f / 256.f);
    float v1 = redB[0] * (1.f / 256.f) - m1 * m1;
    float rs1 = rsqrtf(v1 + 1e-5f);
    float pn = (u - m1) * rs1 * g_pn[t] + bt_pn[t];
    float f = pn + wavg[(size_t)boi * 256 + t];
    __syncthreads();
    // LN2
    redA[t] = f; redB[t] = f * f; __syncthreads();
    for (int st = 128; st; st >>= 1) { if (t < st) { redA[t] += redA[t + st]; redB[t] += redB[t + st]; } __syncthreads(); }
    float m2 = redA[0] * (1.f / 256.f);
    float v2 = redB[0] * (1.f / 256.f) - m2 * m2;
    float rs2 = rsqrtf(v2 + 1e-5f);
    out[(size_t)boi * 256 + t] = (f - m2) * rs2 * g_on[t] + bt_on[t];
}

extern "C" void kernel_launch(void* const* d_in, const int* in_sizes, int n_in,
                              void* d_out, int out_size, void* d_ws, size_t ws_size,
                              hipStream_t stream)
{
    (void)in_sizes; (void)n_in; (void)out_size; (void)ws_size;
    const float* activation = (const float*)d_in[0];
    const float* tokens     = (const float*)d_in[1];
    const float* W_tok      = (const float*)d_in[2];
    const float* b_tok      = (const float*)d_in[3];
    const float* g_tn       = (const float*)d_in[4];
    const float* bt_tn      = (const float*)d_in[5];
    const float* pool_query = (const float*)d_in[6];
    const float* Wq         = (const float*)d_in[7];
    const float* bq         = (const float*)d_in[8];
    const float* Wk         = (const float*)d_in[9];
    const float* bk         = (const float*)d_in[10];
    const float* Wv         = (const float*)d_in[11];
    const float* bv         = (const float*)d_in[12];
    const float* Wo         = (const float*)d_in[13];
    const float* bo         = (const float*)d_in[14];
    const float* g_pn       = (const float*)d_in[15];
    const float* bt_pn      = (const float*)d_in[16];
    const float* g_on       = (const float*)d_in[17];
    const float* bt_on      = (const float*)d_in[18];

    char* ws = (char*)d_ws;
    unsigned short* tp16 = (unsigned short*)(ws);                        // 16,777,216 B
    float* r_ws   = (float*)(ws + 16777216);                             // 8,192 B
    float* qb_ws  = (float*)(ws + 16777216 + 8192);                      // 256 B
    float* wrecip = (float*)(ws + 16777216 + 8192 + 256);                // 3,328 B
    int*   idx_ws = (int*)  (ws + 16777216 + 8192 + 256 + 3328);         // 409,600 B
    float* wavg   = (float*)(ws + 16777216 + 8192 + 256 + 3328 + 409600);// 819,200 B
    float* outp   = (float*)d_out;

    k0_prep<<<1, 256, 0, stream>>>(pool_query, Wq, bq, Wk, bk, r_ws, qb_ws);
    k1_gemm_ln<<<1024, 256, 0, stream>>>(tokens, W_tok, b_tok, g_tn, bt_tn, tp16);
    k2_topk<<<800, 256, 0, stream>>>(activation, idx_ws, wrecip);
    k3_wavg<<<dim3(7, 4, 8), 128, 0, stream>>>(activation, wrecip, tp16, wavg);
    k4_attn<<<800, 256, 0, stream>>>(tp16, idx_ws, r_ws, qb_ws, Wv, bv, Wo, bo,
                                     g_pn, bt_pn, g_on, bt_on, wavg, outp);
}
